// SAModule_80934363725909
// MI455X (gfx1250) — matrix operand order repaired and checked by reading, not compiled, not run on hardware
//
#include <hip/hip_runtime.h>
#include <math.h>

typedef __attribute__((ext_vector_type(16))) _Float16 v16h;
typedef __attribute__((ext_vector_type(8)))  float    v8f;

#define NB    16
#define NP    4096
#define KPER  1024
#define MAXN  64
#define CIN   64
#define HDIM  128
#define FK    96      // 67 padded to 96 (3 k-steps of 32)
#define R2    0.04f
#define CAND  1024
#define FSTR  98      // 196B rows -> 49 dwords (odd) -> conflict-free ds loads
#define HSTR  130     // 260B rows -> 65 dwords (odd)

// swizzled weight sizes (halves): [kt][n][hi][e] -> ((kt*128+n)*2+hi)*16+e
#define W1S_ELEMS (3 * HDIM * 2 * 16)   // 12288
#define W2S_ELEMS (4 * HDIM * 2 * 16)   // 16384

__device__ inline unsigned long long shfl_xor_u64(unsigned long long v, int m) {
    unsigned int lo = (unsigned int)v;
    unsigned int hi = (unsigned int)(v >> 32);
    lo = __shfl_xor(lo, m, 32);
    hi = __shfl_xor(hi, m, 32);
    return ((unsigned long long)hi << 32) | lo;
}

// ---------------------------------------------------------------------------
// Kernel 0: convert W1/W2 to f16 *pre-swizzled into WMMA B-fragment order*:
//   Wswz[((kt*128 + n)*2 + hi)*16 + e] = W[(kt*32 + hi*8 + (e&7) + (e>>3)*16)*128 + n]
// so each lane's 16 B-halves are one contiguous 32B chunk (2x b128 loads).
// W1 rows >= 67 are zero padding.
// ---------------------------------------------------------------------------
__global__ void convert_weights_k(const float* __restrict__ W1,
                                  const float* __restrict__ W2,
                                  _Float16* __restrict__ W1s,
                                  _Float16* __restrict__ W2s) {
    const int i = blockIdx.x * blockDim.x + threadIdx.x;
    if (i < W1S_ELEMS) {
        const int e  = i & 15;
        const int hi = (i >> 4) & 1;
        const int n  = (i >> 5) & 127;
        const int kt = i >> 12;
        const int row = kt * 32 + (e & 7) + hi * 8 + ((e >> 3) << 4);
        W1s[i] = (row < CIN + 3) ? (_Float16)W1[row * HDIM + n] : (_Float16)0.0f;
    } else if (i < W1S_ELEMS + W2S_ELEMS) {
        const int j  = i - W1S_ELEMS;
        const int e  = j & 15;
        const int hi = (j >> 4) & 1;
        const int n  = (j >> 5) & 127;
        const int kt = j >> 12;
        const int row = kt * 32 + (e & 7) + hi * 8 + ((e >> 3) << 4);
        W2s[j] = (_Float16)W2[row * HDIM + n];
    }
}

// ---------------------------------------------------------------------------
// Kernel 1: farthest point sampling. 1 block / cloud, 512 threads (16 waves).
// Positions in LDS; per-thread min_d2 in registers (8 pts/thread).
// argmax via packed (f32bits<<32 | ~idx) u64 max -> JAX first-index tiebreak.
// Writes new_pos directly (== centers).
// ---------------------------------------------------------------------------
__global__ __launch_bounds__(512)
void fps_k(const float* __restrict__ pos, float* __restrict__ newpos) {
    __shared__ float px[NP], py[NP], pz[NP];
    __shared__ unsigned long long warr[16];
    __shared__ int sel_s;

    const int b    = blockIdx.x;
    const int tid  = threadIdx.x;
    const int lane = tid & 31;
    const int wv   = tid >> 5;

    const float* p = pos + (size_t)b * NP * 3;
    for (int i = tid; i < NP; i += 512) {
        px[i] = p[i * 3 + 0];
        py[i] = p[i * 3 + 1];
        pz[i] = p[i * 3 + 2];
    }
    if (tid == 0) sel_s = 0;

    float md[8];
#pragma unroll
    for (int i = 0; i < 8; ++i) md[i] = INFINITY;
    __syncthreads();

    float* np_out = newpos + (size_t)b * KPER * 3;
    for (int k = 0; k < KPER; ++k) {
        const int s = sel_s;
        if (tid == 0) {
            np_out[k * 3 + 0] = px[s];
            np_out[k * 3 + 1] = py[s];
            np_out[k * 3 + 2] = pz[s];
        }
        const float sx = px[s], sy = py[s], sz = pz[s];
        unsigned long long key = 0ull;
#pragma unroll
        for (int i = 0; i < 8; ++i) {
            const int pi = tid + 512 * i;
            const float dx = px[pi] - sx, dy = py[pi] - sy, dz = pz[pi] - sz;
            const float d2 = dx * dx + dy * dy + dz * dz;
            const float m  = fminf(md[i], d2);
            md[i] = m;
            const unsigned long long kk =
                ((unsigned long long)__float_as_uint(m) << 32) |
                (unsigned int)(0xFFFFFFFFu ^ (unsigned int)pi);
            key = (kk > key) ? kk : key;
        }
#pragma unroll
        for (int off = 16; off > 0; off >>= 1) {
            const unsigned long long o = shfl_xor_u64(key, off);
            key = (o > key) ? o : key;
        }
        if (lane == 0) warr[wv] = key;
        __syncthreads();
        if (wv == 0) {
            unsigned long long kk = (lane < 16) ? warr[lane] : 0ull;
#pragma unroll
            for (int off = 16; off > 0; off >>= 1) {
                const unsigned long long o = shfl_xor_u64(kk, off);
                kk = (o > kk) ? o : kk;
            }
            if (lane == 0)
                sel_s = (int)(0xFFFFFFFFu ^ (unsigned int)(kk & 0xFFFFFFFFull));
        }
        __syncthreads();
    }
}

// ---------------------------------------------------------------------------
// Kernel 2 (fused): radius search + exact 64-nearest-in-ball + PointConv MLP
// with f16 WMMA + masked max aggregation. 1 block (256 thr, 8 waves) / center.
// Wave w owns output cols [16w,16w+16) across all 4 M-tiles; kt-outer loop
// loads each pre-swizzled B fragment once (2x b128) for 4 accumulating WMMAs.
// ---------------------------------------------------------------------------
__global__ __launch_bounds__(256)
void pointconv_k(const float* __restrict__ x, const float* __restrict__ pos,
                 const _Float16* __restrict__ W1s, const float* __restrict__ b1,
                 const _Float16* __restrict__ W2s, const float* __restrict__ b2,
                 const float* __restrict__ newpos, float* __restrict__ out) {
    __shared__ unsigned long long cand[CAND];
    __shared__ int nidx[MAXN];
    __shared__ _Float16 feat[64][FSTR];
    __shared__ _Float16 h1[64][HSTR];
    __shared__ int cnt_s;
    __shared__ float cs[3];

    const int g    = blockIdx.x;      // center id
    const int b    = g >> 10;         // cloud id (KPER = 1024)
    const int base = b * NP;
    const int tid  = threadIdx.x;

#pragma unroll
    for (int r = 0; r < 4; ++r) cand[tid + 256 * r] = ~0ull;
    if (tid == 0) cnt_s = 0;
    if (tid < 3) cs[tid] = newpos[g * 3 + tid];
    __syncthreads();

    // ---- phase A: collect within-radius candidates (d2bits<<32 | idx) ----
    const float cx = cs[0], cy = cs[1], cz = cs[2];
    for (int i = tid; i < NP; i += 256) {
        const float* pp = pos + (size_t)(base + i) * 3;
        const float dx = pp[0] - cx;
        const float dy = pp[1] - cy;
        const float dz = pp[2] - cz;
        const float d2 = dx * dx + dy * dy + dz * dz;
        if (d2 <= R2) {
            const int p = atomicAdd(&cnt_s, 1);
            if (p < CAND)
                cand[p] = ((unsigned long long)__float_as_uint(d2) << 32) |
                          (unsigned int)i;
        }
    }
    __syncthreads();
    const int cnt = cnt_s;
    const int m   = (cnt < MAXN) ? cnt : MAXN;

    // ---- sort only when truncation matters (exact 64 nearest) ----
    if (cnt > MAXN) {
        for (int ksz = 2; ksz <= CAND; ksz <<= 1) {
            for (int j = ksz >> 1; j > 0; j >>= 1) {
#pragma unroll
                for (int r = 0; r < 4; ++r) {
                    const int i = tid + 256 * r;
                    const int l = i ^ j;
                    if (l > i) {
                        const unsigned long long a = cand[i], bb = cand[l];
                        const bool up = ((i & ksz) == 0);
                        if ((a > bb) == up) { cand[i] = bb; cand[l] = a; }
                    }
                }
                __syncthreads();
            }
        }
    }
    if (tid < MAXN)
        nidx[tid] = (tid < m) ? (int)(unsigned int)(cand[tid] & 0xFFFFFFFFull) : 0;
    __syncthreads();

    // ---- phase B: build feat[64][96] = [x_j | pos_j - c | 0-pad] in f16 ----
    {
        const int row = tid >> 2, sub = tid & 3;
        const int c0  = sub * 24;
        if (row < m) {
            const int gi = base + nidx[row];
#pragma unroll
            for (int c = 0; c < 24; ++c) {
                const int col = c0 + c;
                float v;
                if (col < CIN)          v = x[(size_t)gi * CIN + col];
                else if (col < CIN + 3) v = pos[(size_t)gi * 3 + (col - CIN)] - cs[col - CIN];
                else                    v = 0.0f;
                feat[row][col] = (_Float16)v;
            }
        } else {
#pragma unroll
            for (int c = 0; c < 24; ++c) feat[row][c0 + c] = (_Float16)0.0f;
        }
    }
    __syncthreads();

    const int lane = tid & 31;
    const int wv   = tid >> 5;       // 0..7
    const int hi   = lane >> 4;      // half-lane group
    const int l15  = lane & 15;
    const int n0   = wv * 16;
    const int n    = n0 + l15;       // output column this lane owns

    // ---- layer 1: h1 = relu(feat @ W1 + b1), K = 96 (3 wmma k-steps) ----
    {
        const float bias = b1[n];
        v8f acc[4];
#pragma unroll
        for (int mt = 0; mt < 4; ++mt)
#pragma unroll
            for (int j = 0; j < 8; ++j) acc[mt][j] = bias;

#pragma unroll
        for (int kt = 0; kt < 3; ++kt) {
            const int k0 = kt * 32;
            // pre-swizzled B fragment: one contiguous 32B chunk per lane
            const v16h bf = *(const v16h*)(W1s + (size_t)((kt * 128 + n) * 2 + hi) * 16);
#pragma unroll
            for (int mt = 0; mt < 4; ++mt) {
                const int m0 = mt * 16;
                v16h a;
#pragma unroll
                for (int e = 0; e < 16; ++e) {
                    // A 16x32 f16 layout: lanes 0-15 K={0..7,16..23}, lanes 16-31 +8
                    const int kk = (e & 7) + ((e >> 3) << 4) + hi * 8;
                    a[e] = feat[m0 + l15][k0 + kk];
                }
                acc[mt] = __builtin_amdgcn_wmma_f32_16x16x32_f16(
                    false, a, false, bf, (short)0, acc[mt], false, false);
            }
        }
#pragma unroll
        for (int mt = 0; mt < 4; ++mt)
#pragma unroll
            for (int j = 0; j < 8; ++j)   // D layout: row = m0 + j + 8*hi
                h1[mt * 16 + j + hi * 8][n] = (_Float16)fmaxf(acc[mt][j], 0.0f);
    }
    __syncthreads();

    // ---- layer 2 + masked max: K = 128 (4 wmma k-steps) ----
    {
        const float bias = b2[n];
        v8f acc[4];
#pragma unroll
        for (int mt = 0; mt < 4; ++mt)
#pragma unroll
            for (int j = 0; j < 8; ++j) acc[mt][j] = bias;

#pragma unroll
        for (int kt = 0; kt < 4; ++kt) {
            const int k0 = kt * 32;
            const v16h bf = *(const v16h*)(W2s + (size_t)((kt * 128 + n) * 2 + hi) * 16);
#pragma unroll
            for (int mt = 0; mt < 4; ++mt) {
                const int m0 = mt * 16;
                v16h a;
#pragma unroll
                for (int e = 0; e < 16; ++e) {
                    const int kk = (e & 7) + ((e >> 3) << 4) + hi * 8;
                    a[e] = h1[m0 + l15][k0 + kk];
                }
                acc[mt] = __builtin_amdgcn_wmma_f32_16x16x32_f16(
                    false, a, false, bf, (short)0, acc[mt], false, false);
            }
        }

        float vmax = -INFINITY;
#pragma unroll
        for (int mt = 0; mt < 4; ++mt) {
#pragma unroll
            for (int j = 0; j < 8; ++j) {
                const int row = mt * 16 + j + hi * 8;
                const float v = fmaxf(acc[mt][j], 0.0f);
                if (row < m) vmax = fmaxf(vmax, v);
            }
        }
        vmax = fmaxf(vmax, __shfl_xor(vmax, 16, 32));  // merge the two row halves
        if (lane < 16) out[(size_t)g * HDIM + n] = vmax;
    }
}

// ---------------------------------------------------------------------------
// Kernel 3: new_batch tail (float or int64 slots, decided from out_size)
// ---------------------------------------------------------------------------
__global__ void write_batch_k(float* tf, long long* ti, int mode) {
    const int i = blockIdx.x * blockDim.x + threadIdx.x;
    if (i < NB * KPER) {
        const int bb = i >> 10;
        if (mode) ti[i] = (long long)bb;
        else      tf[i] = (float)bb;
    }
}

extern "C" void kernel_launch(void* const* d_in, const int* in_sizes, int n_in,
                              void* d_out, int out_size, void* d_ws, size_t ws_size,
                              hipStream_t stream) {
    const float* x   = (const float*)d_in[0];
    const float* pos = (const float*)d_in[1];
    // d_in[2] = batch (unused: layout is fixed/sorted)
    const float* W1  = (const float*)d_in[3];
    const float* b1  = (const float*)d_in[4];
    const float* W2  = (const float*)d_in[5];
    const float* b2  = (const float*)d_in[6];

    float* out    = (float*)d_out;
    float* newpos = out + (size_t)NB * KPER * HDIM;   // new_pos region of d_out

    char* ws = (char*)d_ws;
    _Float16* W1s = (_Float16*)ws;                                   // 24576 B
    _Float16* W2s = (_Float16*)(ws + (size_t)W1S_ELEMS * sizeof(_Float16)); // 32B-aligned

    const int conv_total = W1S_ELEMS + W2S_ELEMS;
    convert_weights_k<<<(conv_total + 255) / 256, 256, 0, stream>>>(W1, W2, W1s, W2s);

    fps_k<<<NB, 512, 0, stream>>>(pos, newpos);

    pointconv_k<<<NB * KPER, 256, 0, stream>>>(x, pos, W1s, b1, W2s, b2, newpos, out);

    const long long tail =
        (long long)out_size - (long long)(NB * KPER * HDIM + NB * KPER * 3);
    if (tail >= NB * KPER) {
        const int mode = (tail >= 2LL * NB * KPER) ? 1 : 0;  // int64 occupies 2 slots
        float* tf = newpos + (size_t)NB * KPER * 3;
        write_batch_k<<<(NB * KPER + 255) / 256, 256, 0, stream>>>(
            tf, (long long*)tf, mode);
    }
}
